// GATDismantlingLayer_56135222559299
// MI455X (gfx1250) — matrix operand order repaired and checked by reading, not compiled, not run on hardware
//
#include <hip/hip_runtime.h>
#include <hip/hip_bf16.h>
#include <math.h>

// ---------------------------------------------------------------------------
// GAT dismantling network, 3 layers, for MI455X (gfx1250, wave32, WMMA).
// GEMMs run as bf16 WMMA 16x16x32 with f32 accum. Each 256-thread block (8
// waves) shares a 4-n-tile B strip staged in LDS (split-half layout to avoid
// bank conflicts); each wave computes a 16x64 strip with A double-buffered
// from global. Attention softmax/scatter is fp32 VALU with float atomics.
// ---------------------------------------------------------------------------

#define LEAKY_SLOPE 0.2f

typedef __attribute__((ext_vector_type(16))) __bf16 v16bf;
typedef __attribute__((ext_vector_type(8)))  float  v8f;

__device__ __forceinline__ unsigned short f32_bf16(float f) {
  unsigned int u = __float_as_uint(f);
  u += 0x7FFFu + ((u >> 16) & 1u);          // round-to-nearest-even
  return (unsigned short)(u >> 16);
}

// ---- pack A (activations, row-major [N,K] f32) into WMMA A-matrix layout ----
// slot index = ((mt*KB + kt)*32 + lane)*16 + e
// lane<16 : M = mt*16+lane,    e0..7 -> K = kt*32+e,    e8..15 -> K = kt*32+16+(e-8)
// lane>=16: M = mt*16+lane-16, e0..7 -> K = kt*32+8+e,  e8..15 -> K = kt*32+24+(e-8)
__global__ void k_pack_a(const float* __restrict__ src, unsigned short* __restrict__ dst,
                         int Mtiles, int KB, int K) {
  int idx = blockIdx.x * blockDim.x + threadIdx.x;
  int total = Mtiles * KB * 32;
  if (idx >= total) return;
  int lane = idx & 31;
  int kt   = (idx >> 5) % KB;
  int mt   = idx / (KB * 32);
  int half = lane >> 4;
  int row  = mt * 16 + (lane & 15);
  int kb   = kt * 32 + half * 8;
  const float* sp = src + (size_t)row * K;
  unsigned short* dp = dst + (size_t)idx * 16;
#pragma unroll
  for (int e = 0; e < 8; ++e) dp[e]     = f32_bf16(sp[kb + e]);
#pragma unroll
  for (int e = 0; e < 8; ++e) dp[8 + e] = f32_bf16(sp[kb + 16 + e]);
}

// ---- pack B (weights, row-major [K,Nout] f32) into WMMA B-matrix layout ----
// lane l holds col N = nt*16 + (l&15); h=l>>4: e0..7 -> K = kt*32+8h+e, e8..15 -> K = kt*32+16+8h+(e-8)
__global__ void k_pack_b(const float* __restrict__ W, unsigned short* __restrict__ dst,
                         int Ntiles, int KB, int Nout) {
  int idx = blockIdx.x * blockDim.x + threadIdx.x;
  int total = Ntiles * KB * 32;
  if (idx >= total) return;
  int lane = idx & 31;
  int kt   = (idx >> 5) % KB;
  int nt   = idx / (KB * 32);
  const float* sp = W + (size_t)(kt * 32 + (lane >> 4) * 8) * Nout + nt * 16 + (lane & 15);
  unsigned short* dp = dst + (size_t)idx * 16;
#pragma unroll
  for (int e = 0; e < 8; ++e) dp[e] = f32_bf16(sp[(size_t)e * Nout]);
#pragma unroll
  for (int e = 0; e < 8; ++e) dp[8 + e] = f32_bf16(sp[(size_t)(16 + e) * Nout]);
}

union V16U { uint4 q[2]; v16bf v; };

// ---- WMMA GEMM: block = 8 waves sharing one 4-n-tile B strip in LDS ----
// wave w computes rows of m-tile (mtg*8 + w); B strip staged once per block.
template <int KB>
__global__ void __launch_bounds__(256) k_gemm_wmma(
    const unsigned short* __restrict__ Apack,
    const unsigned short* __restrict__ Bpack,
    float* __restrict__ C,
    int Mtiles, int NtilesDiv4, int Nout) {
  // split-half LDS layout: each v16bf slot -> 16B in sB0 + 16B in sB1
  __shared__ uint4 sB0[4 * KB * 32];
  __shared__ uint4 sB1[4 * KB * 32];

  int ns   = blockIdx.x % NtilesDiv4;       // strip of 4 consecutive n-tiles
  int mtg  = blockIdx.x / NtilesDiv4;       // group of 8 m-tiles
  int wid  = threadIdx.x >> 5;
  int lane = threadIdx.x & 31;
  int mt   = mtg * 8 + wid;
  bool active = (mt < Mtiles);
  int mtc  = active ? mt : (Mtiles - 1);    // clamp: safe loads, store guarded

  // ---- cooperative stage of B strip into LDS (contiguous in Bpack) ----
  const int totalSlots = 4 * KB * 32;
  const uint4* gB = reinterpret_cast<const uint4*>(Bpack) + (size_t)(ns * 4) * KB * 32 * 2;
  for (int s = threadIdx.x; s < totalSlots; s += 256) {
    sB0[s] = gB[2 * s];
    sB1[s] = gB[2 * s + 1];
  }
  __syncthreads();

  const v16bf* A = reinterpret_cast<const v16bf*>(Apack) + (size_t)mtc * KB * 32 + lane;

  v8f acc0 = {}, acc1 = {}, acc2 = {}, acc3 = {};
  v16bf a = A[0];

#pragma unroll
  for (int kt = 0; kt < KB; ++kt) {
    v16bf an{};
    if (kt + 1 < KB) an = A[(size_t)(kt + 1) * 32];   // A double-buffered

    int base = kt * 32 + lane;
    V16U u0, u1, u2, u3;
    u0.q[0] = sB0[base];               u0.q[1] = sB1[base];
    u1.q[0] = sB0[base + KB * 32];     u1.q[1] = sB1[base + KB * 32];
    u2.q[0] = sB0[base + 2 * KB * 32]; u2.q[1] = sB1[base + 2 * KB * 32];
    u3.q[0] = sB0[base + 3 * KB * 32]; u3.q[1] = sB1[base + 3 * KB * 32];

    acc0 = __builtin_amdgcn_wmma_f32_16x16x32_bf16(false, a, false, u0.v, (short)0, acc0, false, false);
    acc1 = __builtin_amdgcn_wmma_f32_16x16x32_bf16(false, a, false, u1.v, (short)0, acc1, false, false);
    acc2 = __builtin_amdgcn_wmma_f32_16x16x32_bf16(false, a, false, u2.v, (short)0, acc2, false, false);
    acc3 = __builtin_amdgcn_wmma_f32_16x16x32_bf16(false, a, false, u3.v, (short)0, acc3, false, false);
    a = an;
  }

  if (!active) return;                 // wave-uniform, after all barriers/WMMA

  // C/D layout: VGPR r, lanes 0-15 -> (M=r, N=lane); lanes 16-31 -> (M=8+r, N=lane-16)
  int col0 = ns * 64 + (lane & 15);
  int row0 = mt * 16 + ((lane >> 4) << 3);
  float* Cp = C + (size_t)row0 * Nout + col0;
#pragma unroll
  for (int r = 0; r < 8; ++r) {
    Cp[(size_t)r * Nout]      = acc0[r];
    Cp[(size_t)r * Nout + 16] = acc1[r];
    Cp[(size_t)r * Nout + 32] = acc2[r];
    Cp[(size_t)r * Nout + 48] = acc3[r];
  }
}

// ---- per-node attention logits: alpha_s[n,h] = <h[n,h,:], a_s[h,:]> ----
__global__ void k_alpha(const float* __restrict__ h, const float* __restrict__ avs,
                        const float* __restrict__ avd, float* __restrict__ alpha_s,
                        float* __restrict__ alpha_d, int N, int H, int Cc) {
  int i = blockIdx.x * blockDim.x + threadIdx.x;
  if (i >= N * H) return;
  int n = i / H, hh = i % H;
  const float* hp = h + (size_t)n * H * Cc + (size_t)hh * Cc;
  const float* sp = avs + (size_t)hh * Cc;
  const float* dp = avd + (size_t)hh * Cc;
  float s1 = 0.f, s2 = 0.f;
  for (int c = 0; c < Cc; ++c) { float v = hp[c]; s1 += v * sp[c]; s2 += v * dp[c]; }
  alpha_s[i] = s1;
  alpha_d[i] = s2;
}

__global__ void k_fill(float* __restrict__ p, float v, long long n) {
  long long i = (long long)blockIdx.x * blockDim.x + threadIdx.x;
  if (i < n) p[i] = v;
}

__device__ __forceinline__ void edge_sd(const long long* __restrict__ ei, int E, int e,
                                        int& s, int& d) {
  if (e < E) { s = (int)ei[e]; d = (int)ei[(size_t)E + e]; }
  else       { s = d = e - E; }                 // self loops
}

__device__ __forceinline__ float edge_logit(const float* __restrict__ as,
                                            const float* __restrict__ ad,
                                            int s, int d, int h, int H) {
  float v = as[(size_t)s * H + h] + ad[(size_t)d * H + h];
  return v > 0.f ? v : LEAKY_SLOPE * v;
}

__device__ __forceinline__ void atomicMaxF32(float* addr, float val) {
  if (val >= 0.f) atomicMax((int*)addr, __float_as_int(val));
  else            atomicMin((unsigned int*)addr, __float_as_uint(val));
}

__global__ void k_edge_max(const long long* __restrict__ ei, int E, int Etot, int H,
                           const float* __restrict__ as, const float* __restrict__ ad,
                           float* __restrict__ m) {
  int i = blockIdx.x * blockDim.x + threadIdx.x;
  if (i >= Etot * H) return;
  int e = i / H, h = i % H, s, d;
  edge_sd(ei, E, e, s, d);
  atomicMaxF32(&m[(size_t)d * H + h], edge_logit(as, ad, s, d, h, H));
}

__global__ void k_edge_sum(const long long* __restrict__ ei, int E, int Etot, int H,
                           const float* __restrict__ as, const float* __restrict__ ad,
                           const float* __restrict__ m, float* __restrict__ ssum) {
  int i = blockIdx.x * blockDim.x + threadIdx.x;
  if (i >= Etot * H) return;
  int e = i / H, h = i % H, s, d;
  edge_sd(ei, E, e, s, d);
  float v = edge_logit(as, ad, s, d, h, H);
  atomicAdd(&ssum[(size_t)d * H + h], expf(v - m[(size_t)d * H + h]));
}

__global__ void k_edge_scatter(const long long* __restrict__ ei, int E, int Etot, int H, int Cc,
                               const float* __restrict__ as, const float* __restrict__ ad,
                               const float* __restrict__ m, const float* __restrict__ ssum,
                               const float* __restrict__ hfeat, float* __restrict__ out) {
  int i = blockIdx.x * blockDim.x + threadIdx.x;
  if (i >= Etot * H) return;
  int e = i / H, h = i % H, s, d;
  edge_sd(ei, E, e, s, d);
  float v = edge_logit(as, ad, s, d, h, H);
  float w = expf(v - m[(size_t)d * H + h]) / (ssum[(size_t)d * H + h] + 1e-16f);
  const float* hp = hfeat + (size_t)s * H * Cc + (size_t)h * Cc;
  float* op = out + (size_t)d * H * Cc + (size_t)h * Cc;
  for (int c = 0; c < Cc; ++c) atomicAdd(&op[c], w * hp[c]);
}

// out[i] = elu(out[i] + b[i % HC])   (in place)
__global__ void k_bias_elu(float* __restrict__ out, const float* __restrict__ b,
                           int N, int HC) {
  long long i = (long long)blockIdx.x * blockDim.x + threadIdx.x;
  if (i >= (long long)N * HC) return;
  float x = out[i] + b[i % HC];
  out[i] = x > 0.f ? x : (expf(x) - 1.f);
}

// layer-3 GEMV + its attention logits (h3 is a scalar per node)
__global__ void k_gemv3(const float* __restrict__ act, const float* __restrict__ W3,
                        const float* __restrict__ a3s, const float* __restrict__ a3d,
                        float* __restrict__ h3, float* __restrict__ as3,
                        float* __restrict__ ad3, int N, int K) {
  int n = blockIdx.x * blockDim.x + threadIdx.x;
  if (n >= N) return;
  const float* ap = act + (size_t)n * K;
  float acc = 0.f;
  for (int k = 0; k < K; ++k) acc += ap[k] * W3[k];
  h3[n]  = acc;
  as3[n] = acc * a3s[0];
  ad3[n] = acc * a3d[0];
}

__global__ void k_final_sigmoid(const float* __restrict__ o3, const float* __restrict__ b3,
                                float* __restrict__ out, int N) {
  int n = blockIdx.x * blockDim.x + threadIdx.x;
  if (n >= N) return;
  float x = o3[n] + b3[0];
  out[n] = 1.f / (1.f + expf(-x));
}

// ---------------------------------------------------------------------------

static inline int cdiv(long long a, long long b) { return (int)((a + b - 1) / b); }

struct LayerBufs {
  unsigned short *Apack, *Wpack;
  float *hbuf, *obuf, *as, *ad, *m, *s;
};

// runs GEMM + attention for one 8-head/64-ch layer (A already packed)
static void gat_layer(const LayerBufs& L, const float* W, const float* avs,
                      const float* avd, const long long* ei,
                      int N, int E, int Etot, int Kin, hipStream_t stream) {
  const int H = 8, Cc = 64, HC = 512;
  const int Mt = N / 16, Nt = HC / 16, KB = Kin / 32;

  k_pack_b<<<cdiv((long long)Nt * KB * 32, 256), 256, 0, stream>>>(W, L.Wpack, Nt, KB, HC);

  int grid = cdiv(Mt, 8) * (Nt / 4);   // block = 8 waves x one B strip
  if (KB == 4)
    k_gemm_wmma<4><<<grid, 256, 0, stream>>>(L.Apack, L.Wpack, L.hbuf, Mt, Nt / 4, HC);
  else
    k_gemm_wmma<16><<<grid, 256, 0, stream>>>(L.Apack, L.Wpack, L.hbuf, Mt, Nt / 4, HC);

  k_alpha<<<cdiv((long long)N * H, 256), 256, 0, stream>>>(L.hbuf, avs, avd, L.as, L.ad, N, H, Cc);

  k_fill<<<cdiv((long long)N * H, 256), 256, 0, stream>>>(L.m, -INFINITY, (long long)N * H);
  k_fill<<<cdiv((long long)N * H, 256), 256, 0, stream>>>(L.s, 0.f, (long long)N * H);
  k_fill<<<cdiv((long long)N * HC, 256), 256, 0, stream>>>(L.obuf, 0.f, (long long)N * HC);

  int EH = Etot * H;
  k_edge_max    <<<cdiv(EH, 256), 256, 0, stream>>>(ei, E, Etot, H, L.as, L.ad, L.m);
  k_edge_sum    <<<cdiv(EH, 256), 256, 0, stream>>>(ei, E, Etot, H, L.as, L.ad, L.m, L.s);
  k_edge_scatter<<<cdiv(EH, 256), 256, 0, stream>>>(ei, E, Etot, H, Cc, L.as, L.ad, L.m, L.s,
                                                    L.hbuf, L.obuf);
}

extern "C" void kernel_launch(void* const* d_in, const int* in_sizes, int n_in,
                              void* d_out, int out_size, void* d_ws, size_t ws_size,
                              hipStream_t stream) {
  const float*     x   = (const float*)d_in[0];
  const long long* ei  = (const long long*)d_in[1];   // int64 [2,E]
  const float*     W1  = (const float*)d_in[2];
  const float*     a1s = (const float*)d_in[3];
  const float*     a1d = (const float*)d_in[4];
  const float*     b1  = (const float*)d_in[5];
  const float*     W2  = (const float*)d_in[6];
  const float*     a2s = (const float*)d_in[7];
  const float*     a2d = (const float*)d_in[8];
  const float*     b2  = (const float*)d_in[9];
  const float*     W3  = (const float*)d_in[10];
  const float*     a3s = (const float*)d_in[11];
  const float*     a3d = (const float*)d_in[12];
  const float*     b3  = (const float*)d_in[13];
  float* out = (float*)d_out;

  const int INDIM = 128, HC = 512;
  const int N    = in_sizes[0] / INDIM;     // 20000 (multiple of 16)
  const int E    = in_sizes[1] / 2;         // 320000
  const int Etot = E + N;                   // + self loops
  const int Mt   = N / 16;

  // ---- carve workspace ----
  char* p = (char*)d_ws;
  auto alloc = [&](size_t bytes) -> char* {
    char* r = p; p += (bytes + 255) & ~(size_t)255; return r;
  };
  LayerBufs L;
  L.hbuf  = (float*)alloc((size_t)N * HC * 4);
  L.obuf  = (float*)alloc((size_t)N * HC * 4);
  L.Apack = (unsigned short*)alloc((size_t)N * HC * 2);              // max K = 512
  L.Wpack = (unsigned short*)alloc((size_t)(HC / 16) * (HC / 32) * 32 * 16 * 2);
  L.as    = (float*)alloc((size_t)N * 8 * 4);
  L.ad    = (float*)alloc((size_t)N * 8 * 4);
  L.m     = (float*)alloc((size_t)N * 8 * 4);
  L.s     = (float*)alloc((size_t)N * 8 * 4);
  float* h3  = (float*)alloc((size_t)N * 4);
  float* as3 = (float*)alloc((size_t)N * 4);
  float* ad3 = (float*)alloc((size_t)N * 4);
  float* m3  = (float*)alloc((size_t)N * 4);
  float* s3  = (float*)alloc((size_t)N * 4);
  float* o3  = (float*)alloc((size_t)N * 4);
  (void)ws_size; (void)n_in; (void)out_size;

  // ---------------- layer 1: x[N,128] -> 8x64 ----------------
  {
    int KB = INDIM / 32;
    k_pack_a<<<cdiv((long long)Mt * KB * 32, 256), 256, 0, stream>>>(x, L.Apack, Mt, KB, INDIM);
    gat_layer(L, W1, a1s, a1d, ei, N, E, Etot, INDIM, stream);
    k_bias_elu<<<cdiv((long long)N * HC, 256), 256, 0, stream>>>(L.obuf, b1, N, HC);
  }

  // ---------------- layer 2: act1[N,512] -> 8x64 ----------------
  {
    int KB = HC / 32;
    k_pack_a<<<cdiv((long long)Mt * KB * 32, 256), 256, 0, stream>>>(L.obuf, L.Apack, Mt, KB, HC);
    gat_layer(L, W2, a2s, a2d, ei, N, E, Etot, HC, stream);   // obuf re-zeroed after GEMM
    k_bias_elu<<<cdiv((long long)N * HC, 256), 256, 0, stream>>>(L.obuf, b2, N, HC);
  }

  // ---------------- layer 3: act2[N,512] -> 1x1, sigmoid ----------------
  {
    k_gemv3<<<cdiv(N, 256), 256, 0, stream>>>(L.obuf, W3, a3s, a3d, h3, as3, ad3, N, HC);

    k_fill<<<cdiv(N, 256), 256, 0, stream>>>(m3, -INFINITY, N);
    k_fill<<<cdiv(N, 256), 256, 0, stream>>>(s3, 0.f, N);
    k_fill<<<cdiv(N, 256), 256, 0, stream>>>(o3, 0.f, N);

    k_edge_max    <<<cdiv(Etot, 256), 256, 0, stream>>>(ei, E, Etot, 1, as3, ad3, m3);
    k_edge_sum    <<<cdiv(Etot, 256), 256, 0, stream>>>(ei, E, Etot, 1, as3, ad3, m3, s3);
    k_edge_scatter<<<cdiv(Etot, 256), 256, 0, stream>>>(ei, E, Etot, 1, 1, as3, ad3, m3, s3,
                                                        h3, o3);

    k_final_sigmoid<<<cdiv(N, 256), 256, 0, stream>>>(o3, b3, out, N);
  }
}
